// TemporalEntityEncoder_40029095199353
// MI455X (gfx1250) — matrix operand order repaired and checked by reading, not compiled, not run on hardware
//
#include <hip/hip_runtime.h>

#define HDIM 256
#define EDIM 64
#define MSG_IN 322           // HDIM + EDIM + 2
#define KPAD 352             // MSG_IN padded to multiple of 32
#define NCLS 6
#define BGRAPHS 8
#define M1STRIDE 264         // padded LDS stride for intermediate (16B-aligned rows)
#define MTILE 32             // edges/nodes per block (2 M-subtiles of 16)

typedef __attribute__((ext_vector_type(16))) __bf16        v16bf;
typedef __attribute__((ext_vector_type(8)))  float         v8f;
typedef __attribute__((ext_vector_type(8)))  unsigned int  v8u;

__device__ __forceinline__ unsigned short f2bf(float f) {
  unsigned int u = __float_as_uint(f);
  u += 0x7FFFu + ((u >> 16) & 1u);          // round-to-nearest-even
  return (unsigned short)(u >> 16);
}

// Load a 16x bf16 fragment as two 16-byte chunks separated by `gap` (in ushorts).
// A-matrix fragment: gap=16 (K offsets {0..7} and {16..23} per lane half).
// B-matrix fragment: gap=16 (same K offsets as A).
__device__ __forceinline__ v16bf ldfrag(const unsigned short* p, int gap) {
  uint4 lo = *reinterpret_cast<const uint4*>(p);
  uint4 hi = *reinterpret_cast<const uint4*>(p + gap);
  v8u u = {lo.x, lo.y, lo.z, lo.w, hi.x, hi.y, hi.z, hi.w};
  return __builtin_bit_cast(v16bf, u);
}

#define WMMA_BF16(a, b, c) \
  __builtin_amdgcn_wmma_f32_16x16x32_bf16(false, (a), false, (b), (short)0, (c), false, false)

// ---------------- weight conversion (f32 row-major -> bf16 transposed) ----------------

__global__ void conv_w1_kernel(const float* __restrict__ W, unsigned short* __restrict__ outT) {
  int idx = blockIdx.x * 256 + threadIdx.x;      // idx = n*KPAD + k
  if (idx >= HDIM * KPAD) return;
  int n = idx / KPAD, k = idx % KPAD;
  outT[idx] = (k < MSG_IN) ? f2bf(W[k * HDIM + n]) : (unsigned short)0;
}

__global__ void conv_sq_kernel(const float* __restrict__ W, unsigned short* __restrict__ outT) {
  int idx = blockIdx.x * 256 + threadIdx.x;      // idx = n*HDIM + k
  if (idx >= HDIM * HDIM) return;
  int n = idx / HDIM, k = idx % HDIM;
  outT[idx] = f2bf(W[k * HDIM + n]);
}

__global__ void fill_kernel(float* __restrict__ p, float v, int n) {
  int i = blockIdx.x * 256 + threadIdx.x;
  if (i < n) p[i] = v;
}

// ---------------- edge message kernel: m = relu(z@W1+b1)@W2+b2, scatter-add to dst ----
// 32 edges per block; 8 waves; each wave owns a 32(M) x 32(N) output tile as a
// 2x2 grid of 16x16 WMMA tiles, so each B fragment feeds two WMMAs.

__global__ __launch_bounds__(256)
void edge_msg_kernel(const float* __restrict__ hin,            // [N, H]
                     const float* __restrict__ efeat,          // [E, EDIM]
                     const int*   __restrict__ etype,
                     const int*   __restrict__ src,
                     const int*   __restrict__ dst,
                     const unsigned short* __restrict__ W1T,   // [H][KPAD] bf16
                     const float* __restrict__ b1,
                     const unsigned short* __restrict__ W2T,   // [H][H] bf16
                     const float* __restrict__ b2,
                     float* __restrict__ msum) {               // [N, H]
  __shared__ unsigned short zA[MTILE * KPAD];
  __shared__ unsigned short m1[MTILE * M1STRIDE];
  __shared__ int sSrc[MTILE], sDst[MTILE];

  const int t = threadIdx.x;
  const int ebase = blockIdx.x * MTILE;

  if (t < MTILE) { sSrc[t] = src[ebase + t]; sDst[t] = dst[ebase + t]; }
  __syncthreads();

  // gather z = [h[src] | e_feat | onehot | zero-pad], converted to bf16
  {
    const int r = t >> 3, c0 = t & 7;              // 8 threads per row, 32 rows
    const int e = ebase + r;
    const int s = sSrc[r];
    const int et = etype[e];
    for (int i = 0; i < KPAD / 8; ++i) {
      int c = c0 + 8 * i;
      float v;
      if (c < HDIM)                  v = hin[s * HDIM + c];
      else if (c < HDIM + EDIM)      v = efeat[e * EDIM + (c - HDIM)];
      else if (c == HDIM + EDIM)     v = (et == 0) ? 1.f : 0.f;
      else if (c == HDIM + EDIM + 1) v = (et == 1) ? 1.f : 0.f;
      else                           v = 0.f;
      zA[r * KPAD + c] = f2bf(v);
    }
  }
  __syncthreads();

  const int wave = t >> 5, lane = t & 31;
  const int lrow = lane & 15, lhi = lane >> 4;
  const int col0 = (wave * 2) * 16 + lrow;
  const int col1 = col0 + 16;

  // GEMM1: [32,KPAD] x [KPAD,H]
  v8f a00 = {}, a01 = {}, a10 = {}, a11 = {};
  for (int kt = 0; kt < KPAD / 32; ++kt) {
    v16bf aM0 = ldfrag(&zA[lrow * KPAD + kt * 32 + lhi * 8], 16);
    v16bf aM1 = ldfrag(&zA[(16 + lrow) * KPAD + kt * 32 + lhi * 8], 16);
    v16bf bN0 = ldfrag(&W1T[col0 * KPAD + kt * 32 + lhi * 8], 16);
    v16bf bN1 = ldfrag(&W1T[col1 * KPAD + kt * 32 + lhi * 8], 16);
    a00 = WMMA_BF16(aM0, bN0, a00);
    a01 = WMMA_BF16(aM0, bN1, a01);
    a10 = WMMA_BF16(aM1, bN0, a10);
    a11 = WMMA_BF16(aM1, bN1, a11);
  }
  const float bias0 = b1[col0], bias1 = b1[col1];
#pragma unroll
  for (int r = 0; r < 8; ++r) {
    int rowm = r + lhi * 8;                 // C layout: VGPR r -> M = r (+8 for upper half)
    m1[rowm * M1STRIDE + col0]        = f2bf(fmaxf(a00[r] + bias0, 0.f));
    m1[rowm * M1STRIDE + col1]        = f2bf(fmaxf(a01[r] + bias1, 0.f));
    m1[(16 + rowm) * M1STRIDE + col0] = f2bf(fmaxf(a10[r] + bias0, 0.f));
    m1[(16 + rowm) * M1STRIDE + col1] = f2bf(fmaxf(a11[r] + bias1, 0.f));
  }
  __syncthreads();

  // GEMM2: [32,H] x [H,H]
  v8f o00 = {}, o01 = {}, o10 = {}, o11 = {};
  for (int kt = 0; kt < HDIM / 32; ++kt) {
    v16bf aM0 = ldfrag(&m1[lrow * M1STRIDE + kt * 32 + lhi * 8], 16);
    v16bf aM1 = ldfrag(&m1[(16 + lrow) * M1STRIDE + kt * 32 + lhi * 8], 16);
    v16bf bN0 = ldfrag(&W2T[col0 * HDIM + kt * 32 + lhi * 8], 16);
    v16bf bN1 = ldfrag(&W2T[col1 * HDIM + kt * 32 + lhi * 8], 16);
    o00 = WMMA_BF16(aM0, bN0, o00);
    o01 = WMMA_BF16(aM0, bN1, o01);
    o10 = WMMA_BF16(aM1, bN0, o10);
    o11 = WMMA_BF16(aM1, bN1, o11);
  }
  const float c2_0 = b2[col0], c2_1 = b2[col1];
#pragma unroll
  for (int r = 0; r < 8; ++r) {
    int rowm = r + lhi * 8;
    int d0 = sDst[rowm];
    int d1 = sDst[16 + rowm];
    atomicAdd(&msum[d0 * HDIM + col0], o00[r] + c2_0);
    atomicAdd(&msum[d0 * HDIM + col1], o01[r] + c2_1);
    atomicAdd(&msum[d1 * HDIM + col0], o10[r] + c2_0);
    atomicAdd(&msum[d1 * HDIM + col1], o11[r] + c2_1);
  }
}

// ---------------- node update: h' = LN(relu(h@selfW + b + msum)) ----------------------

__global__ __launch_bounds__(256)
void node_update_kernel(const float* __restrict__ hin,
                        const float* __restrict__ msum,
                        const unsigned short* __restrict__ SWT,  // [H][H] bf16
                        const float* __restrict__ sb,
                        const float* __restrict__ gamma,
                        const float* __restrict__ beta,
                        float* __restrict__ hout) {
  __shared__ unsigned short zA[MTILE * HDIM];
  __shared__ float sout[MTILE * HDIM];
  __shared__ float red[MTILE][8], red2[MTILE][8];
  __shared__ float smu[MTILE], srstd[MTILE];

  const int t = threadIdx.x;
  const int nbase = blockIdx.x * MTILE;

  {
    const int r = t >> 3, c0 = t & 7;
    for (int i = 0; i < HDIM / 8; ++i) {
      int c = c0 + 8 * i;
      zA[r * HDIM + c] = f2bf(hin[(nbase + r) * HDIM + c]);
    }
  }
  __syncthreads();

  const int wave = t >> 5, lane = t & 31;
  const int lrow = lane & 15, lhi = lane >> 4;
  const int col0 = (wave * 2) * 16 + lrow;
  const int col1 = col0 + 16;

  v8f a00 = {}, a01 = {}, a10 = {}, a11 = {};
  for (int kt = 0; kt < HDIM / 32; ++kt) {
    v16bf aM0 = ldfrag(&zA[lrow * HDIM + kt * 32 + lhi * 8], 16);
    v16bf aM1 = ldfrag(&zA[(16 + lrow) * HDIM + kt * 32 + lhi * 8], 16);
    v16bf bN0 = ldfrag(&SWT[col0 * HDIM + kt * 32 + lhi * 8], 16);
    v16bf bN1 = ldfrag(&SWT[col1 * HDIM + kt * 32 + lhi * 8], 16);
    a00 = WMMA_BF16(aM0, bN0, a00);
    a01 = WMMA_BF16(aM0, bN1, a01);
    a10 = WMMA_BF16(aM1, bN0, a10);
    a11 = WMMA_BF16(aM1, bN1, a11);
  }
  const float sb0 = sb[col0], sb1 = sb[col1];
#pragma unroll
  for (int r = 0; r < 8; ++r) {
    int rowm = r + lhi * 8;
    int n0 = nbase + rowm;
    int n1 = nbase + 16 + rowm;
    sout[rowm * HDIM + col0]        = fmaxf(a00[r] + sb0 + msum[n0 * HDIM + col0], 0.f);
    sout[rowm * HDIM + col1]        = fmaxf(a01[r] + sb1 + msum[n0 * HDIM + col1], 0.f);
    sout[(16 + rowm) * HDIM + col0] = fmaxf(a10[r] + sb0 + msum[n1 * HDIM + col0], 0.f);
    sout[(16 + rowm) * HDIM + col1] = fmaxf(a11[r] + sb1 + msum[n1 * HDIM + col1], 0.f);
  }
  __syncthreads();

  // LayerNorm per row (8 threads/row partial reduce over 32 elems each)
  {
    const int r = t >> 3, p = t & 7;
    float s = 0.f, s2 = 0.f;
    for (int i = 0; i < 32; ++i) {
      float v = sout[r * HDIM + p * 32 + i];
      s += v; s2 += v * v;
    }
    red[r][p] = s; red2[r][p] = s2;
  }
  __syncthreads();
  if ((t & 7) == 0) {
    const int r = t >> 3;
    float s = 0.f, s2 = 0.f;
    for (int i = 0; i < 8; ++i) { s += red[r][i]; s2 += red2[r][i]; }
    float mu = s * (1.f / HDIM);
    float var = s2 * (1.f / HDIM) - mu * mu;
    smu[r] = mu;
    srstd[r] = rsqrtf(var + 1e-5f);
  }
  __syncthreads();
  {
    const int r = t >> 3, p = t & 7;
    const float mu = smu[r], rs = srstd[r];
    const int node = nbase + r;
    for (int i = 0; i < 32; ++i) {
      int c = p + 8 * i;
      hout[node * HDIM + c] = (sout[r * HDIM + c] - mu) * rs * gamma[c] + beta[c];
    }
  }
}

// ---------------- attention pooling + classifier --------------------------------------

__global__ void score_kernel(const float* __restrict__ h, const float* __restrict__ attW,
                             const float* __restrict__ attb, const int* __restrict__ gid,
                             float* __restrict__ score, float* __restrict__ smax, int N) {
  int n = blockIdx.x * 256 + threadIdx.x;
  if (n >= N) return;
  float s = attb[0];
  for (int k = 0; k < HDIM; ++k) s += h[n * HDIM + k] * attW[k];
  score[n] = s;
  float* addr = &smax[gid[n]];
  if (s >= 0.f) atomicMax((int*)addr, __float_as_int(s));
  else          atomicMin((unsigned int*)addr, __float_as_uint(s));
}

__global__ void exp_kernel(const float* __restrict__ score, const float* __restrict__ smax,
                           const int* __restrict__ gid, float* __restrict__ ex,
                           float* __restrict__ den, int N) {
  int n = blockIdx.x * 256 + threadIdx.x;
  if (n >= N) return;
  float e = expf(score[n] - smax[gid[n]]);
  ex[n] = e;
  atomicAdd(&den[gid[n]], e);
}

__global__ void readout_kernel(const float* __restrict__ h, const float* __restrict__ ex,
                               const float* __restrict__ den, const int* __restrict__ gid,
                               float* __restrict__ readout, int N) {
  int idx = blockIdx.x * 256 + threadIdx.x;
  int n = idx >> 8, c = idx & 255;
  if (n >= N) return;
  int g = gid[n];
  float w = ex[n] / den[g];
  atomicAdd(&readout[g * HDIM + c], h[n * HDIM + c] * w);
}

__global__ __launch_bounds__(256)
void cls_kernel(const float* __restrict__ readout,
                const float* __restrict__ W1, const float* __restrict__ b1,
                const float* __restrict__ W2, const float* __restrict__ b2,
                float* __restrict__ out) {
  __shared__ float sr[BGRAPHS * HDIM];
  __shared__ float sh[BGRAPHS * HDIM];
  const int t = threadIdx.x;
  for (int i = t; i < BGRAPHS * HDIM; i += 256) sr[i] = readout[i];
  __syncthreads();
  for (int b = 0; b < BGRAPHS; ++b) {
    float acc = b1[t];
    for (int k = 0; k < HDIM; ++k) acc += sr[b * HDIM + k] * W1[k * HDIM + t];
    sh[b * HDIM + t] = fmaxf(acc, 0.f);
  }
  __syncthreads();
  if (t < BGRAPHS * NCLS) {
    int b = t / NCLS, c = t % NCLS;
    float acc = b2[c];
    for (int k = 0; k < HDIM; ++k) acc += sh[b * HDIM + k] * W2[k * NCLS + c];
    out[t] = acc;
  }
}

// ---------------- launch --------------------------------------------------------------

extern "C" void kernel_launch(void* const* d_in, const int* in_sizes, int n_in,
                              void* d_out, int out_size, void* d_ws, size_t ws_size,
                              hipStream_t stream) {
  const float* feat  = (const float*)d_in[0];
  const float* efeat = (const float*)d_in[1];
  const float* l0W1  = (const float*)d_in[2];
  const float* l0b1  = (const float*)d_in[3];
  const float* l0W2  = (const float*)d_in[4];
  const float* l0b2  = (const float*)d_in[5];
  const float* l0sW  = (const float*)d_in[6];
  const float* l0sb  = (const float*)d_in[7];
  const float* l0g   = (const float*)d_in[8];
  const float* l0bb  = (const float*)d_in[9];
  const float* l1W1  = (const float*)d_in[10];
  const float* l1b1  = (const float*)d_in[11];
  const float* l1W2  = (const float*)d_in[12];
  const float* l1b2  = (const float*)d_in[13];
  const float* l1sW  = (const float*)d_in[14];
  const float* l1sb  = (const float*)d_in[15];
  const float* l1g   = (const float*)d_in[16];
  const float* l1bb  = (const float*)d_in[17];
  const float* attW  = (const float*)d_in[18];
  const float* attb  = (const float*)d_in[19];
  const float* cW1   = (const float*)d_in[20];
  const float* cb1   = (const float*)d_in[21];
  const float* cW2   = (const float*)d_in[22];
  const float* cb2   = (const float*)d_in[23];
  const int* etype   = (const int*)d_in[24];
  const int* src     = (const int*)d_in[25];
  const int* dst     = (const int*)d_in[26];
  const int* gid     = (const int*)d_in[27];
  float* out = (float*)d_out;

  const int N = in_sizes[27];
  const int E = in_sizes[24];

  // workspace layout
  char* ws = (char*)d_ws;
  size_t off = 0;
  auto alloc = [&](size_t bytes) -> void* {
    void* p = ws + off;
    off += (bytes + 255) & ~size_t(255);
    return p;
  };
  float* h0    = (float*)alloc((size_t)N * HDIM * 4);
  float* h1    = (float*)alloc((size_t)N * HDIM * 4);
  float* msum0 = (float*)alloc((size_t)N * HDIM * 4);
  float* msum1 = (float*)alloc((size_t)N * HDIM * 4);
  float* score = (float*)alloc((size_t)N * 4);
  float* ex    = (float*)alloc((size_t)N * 4);
  float* smax  = (float*)alloc(BGRAPHS * 4);
  float* den   = (float*)alloc(BGRAPHS * 4);
  float* rdout = (float*)alloc(BGRAPHS * HDIM * 4);
  unsigned short* w1l0 = (unsigned short*)alloc((size_t)HDIM * KPAD * 2);
  unsigned short* w2l0 = (unsigned short*)alloc((size_t)HDIM * HDIM * 2);
  unsigned short* swl0 = (unsigned short*)alloc((size_t)HDIM * HDIM * 2);
  unsigned short* w1l1 = (unsigned short*)alloc((size_t)HDIM * KPAD * 2);
  unsigned short* w2l1 = (unsigned short*)alloc((size_t)HDIM * HDIM * 2);
  unsigned short* swl1 = (unsigned short*)alloc((size_t)HDIM * HDIM * 2);
  (void)ws_size; (void)n_in; (void)out_size;

  // 1) convert weights to bf16 (transposed, zero-padded K)
  conv_w1_kernel<<<(HDIM * KPAD + 255) / 256, 256, 0, stream>>>(l0W1, w1l0);
  conv_sq_kernel<<<(HDIM * HDIM + 255) / 256, 256, 0, stream>>>(l0W2, w2l0);
  conv_sq_kernel<<<(HDIM * HDIM + 255) / 256, 256, 0, stream>>>(l0sW, swl0);
  conv_w1_kernel<<<(HDIM * KPAD + 255) / 256, 256, 0, stream>>>(l1W1, w1l1);
  conv_sq_kernel<<<(HDIM * HDIM + 255) / 256, 256, 0, stream>>>(l1W2, w2l1);
  conv_sq_kernel<<<(HDIM * HDIM + 255) / 256, 256, 0, stream>>>(l1sW, swl1);

  // 2) per-call init of accumulation buffers
  const int nh = N * HDIM;
  fill_kernel<<<(nh + 255) / 256, 256, 0, stream>>>(msum0, 0.f, nh);
  fill_kernel<<<(nh + 255) / 256, 256, 0, stream>>>(msum1, 0.f, nh);
  fill_kernel<<<1, 256, 0, stream>>>(smax, -INFINITY, BGRAPHS);
  fill_kernel<<<1, 256, 0, stream>>>(den, 0.f, BGRAPHS);
  fill_kernel<<<(BGRAPHS * HDIM + 255) / 256, 256, 0, stream>>>(rdout, 0.f, BGRAPHS * HDIM);

  // 3) layer 0
  edge_msg_kernel<<<E / MTILE, 256, 0, stream>>>(feat, efeat, etype, src, dst,
                                                 w1l0, l0b1, w2l0, l0b2, msum0);
  node_update_kernel<<<N / MTILE, 256, 0, stream>>>(feat, msum0, swl0, l0sb, l0g, l0bb, h0);

  // 4) layer 1
  edge_msg_kernel<<<E / MTILE, 256, 0, stream>>>(h0, efeat, etype, src, dst,
                                                 w1l1, l1b1, w2l1, l1b2, msum1);
  node_update_kernel<<<N / MTILE, 256, 0, stream>>>(h0, msum1, swl1, l1sb, l1g, l1bb, h1);

  // 5) attention pooling (segment softmax) + readout
  score_kernel<<<(N + 255) / 256, 256, 0, stream>>>(h1, attW, attb, gid, score, smax, N);
  exp_kernel<<<(N + 255) / 256, 256, 0, stream>>>(score, smax, gid, ex, den, N);
  readout_kernel<<<(N * HDIM + 255) / 256, 256, 0, stream>>>(h1, ex, den, gid, rdout, N);

  // 6) classifier
  cls_kernel<<<1, 256, 0, stream>>>(rdout, cW1, cb1, cW2, cb2, out);
}
